// ContrastiveLoss_65180423684716
// MI455X (gfx1250) — matrix operand order repaired and checked by reading, not compiled, not run on hardware
//
#include <hip/hip_runtime.h>
#include <hip/hip_bf16.h>

typedef _Float16 half_t;
typedef __attribute__((ext_vector_type(16))) _Float16 v16h;
typedef __attribute__((ext_vector_type(8)))  _Float16 v8h;
typedef __attribute__((ext_vector_type(8)))  float    v8f;
typedef unsigned int uint;
typedef __attribute__((ext_vector_type(4))) uint u32x4;
typedef __attribute__((ext_vector_type(8))) uint u32x8;

// Problem constants
static constexpr int Bn = 128;
static constexpr int Rn = 36;
static constexpr int Wn = 50;
static constexpr int Dn = 1024;

// Tensor Data Mover staging (CDNA5): 1 = tensor_load_to_lds + double buffering,
// 0 = per-lane global->reg->LDS staging.
#define USE_TDM 1

// ---------------------------------------------------------------------------
// Preprocess: f32 -> f16 copy of im
// ---------------------------------------------------------------------------
__global__ void k_prep_im(const float* __restrict__ im, half_t* __restrict__ imh, int n) {
    int idx = blockIdx.x * blockDim.x + threadIdx.x;
    if (idx < n) imh[idx] = (half_t)im[idx];
}

// ---------------------------------------------------------------------------
// Preprocess: cap = s * mask (f16) and w1[b,w] = ||cap[b,w,:]||
// ---------------------------------------------------------------------------
__global__ void k_prep_cap(const float* __restrict__ s, const int* __restrict__ s_l,
                           half_t* __restrict__ caph, float* __restrict__ w1) {
    int bw = blockIdx.x;              // 0..B*W-1
    int b  = bw / Wn, w = bw % Wn;
    bool keep = (w < s_l[b]);
    const float* srow = s + (size_t)bw * Dn;
    half_t*      drow = caph + (size_t)bw * Dn;
    float ss = 0.f;
    for (int d = threadIdx.x; d < Dn; d += blockDim.x) {
        float v = keep ? srow[d] : 0.f;
        drow[d] = (half_t)v;
        ss += v * v;
    }
    for (int off = 16; off > 0; off >>= 1) ss += __shfl_down(ss, off, 32);
    __shared__ float part[8];
    int lane = threadIdx.x & 31, wv = threadIdx.x >> 5;
    if (lane == 0) part[wv] = ss;
    __syncthreads();
    if (threadIdx.x == 0) {
        float t = 0.f;
        int nw = blockDim.x >> 5;
        for (int k = 0; k < nw; ++k) t += part[k];
        w1[bw] = sqrtf(t);
    }
}

// ---------------------------------------------------------------------------
// Preprocess: Gram[i] = im_i * im_i^T  (36x36 f32), one block per i
// ---------------------------------------------------------------------------
__global__ void k_prep_gram(const float* __restrict__ im, float* __restrict__ gram) {
    const int i = blockIdx.x;
    __shared__ float chunk[Rn * 64];
    float acc[6];
#pragma unroll
    for (int q = 0; q < 6; ++q) acc[q] = 0.f;
    for (int k0 = 0; k0 < Dn; k0 += 64) {
        for (int t = threadIdx.x; t < Rn * 64; t += blockDim.x)
            chunk[t] = im[((size_t)i * Rn + (t >> 6)) * Dn + k0 + (t & 63)];
        __syncthreads();
        int q = 0;
        for (int e = threadIdx.x; e < Rn * Rn; e += blockDim.x, ++q) {
            int r = e / Rn, r2 = e % Rn;
            const float* pr = &chunk[r * 64];
            const float* p2 = &chunk[r2 * 64];
            float ssum = 0.f;
#pragma unroll 8
            for (int k = 0; k < 64; ++k) ssum += pr[k] * p2[k];
            acc[q] += ssum;
        }
        __syncthreads();
    }
    int q = 0;
    for (int e = threadIdx.x; e < Rn * Rn; e += blockDim.x, ++q)
        gram[(size_t)i * (Rn * Rn) + e] = acc[q];
}

#if USE_TDM
// ---------------------------------------------------------------------------
// Build a 2D Tensor-DMA descriptor and issue tensor_load_to_lds.
//   Moves `rows` x `tile_k` f16 elements from a row-major tensor with
//   `row_stride` (elements) into LDS starting at byte offset lds_off,
//   with 4 DWORDs of LDS padding after every 64 DWORDs (one 128-half row),
//   i.e. an effective LDS row stride of 136 halves.
// All inputs are wave-uniform -> descriptor lives in SGPRs.
// ---------------------------------------------------------------------------
__device__ __forceinline__ void tdm_load_2d(uint lds_off, const half_t* g,
                                            uint rows, uint row_stride, uint tile_k) {
    unsigned long long ga = (unsigned long long)(size_t)g;
    u32x4 g0;
    g0[0] = 1u;                                              // count=1, user mode
    g0[1] = lds_off;                                         // lds_addr (bytes)
    g0[2] = (uint)ga;                                        // global_addr[31:0]
    g0[3] = (uint)((ga >> 32) & 0x01FFFFFFu) | (2u << 30);   // addr[56:32] | type=2
    u32x8 g1;
    g1[0] = (1u << 16)            // data_size = 1 -> 2 bytes
          | (1u << 20)            // pad_enable
          | (5u << 22)            // pad_interval: 64 DWORDs
          | (3u << 25);           // pad_amount: 4 DWORDs
    g1[1] = (row_stride & 0xFFFFu) << 16;        // tensor_dim0 low16 (=1024)
    g1[2] = (rows << 16);                        // dim0 hi16=0 | tensor_dim1 low16
    g1[3] = (tile_k << 16);                      // dim1 hi16=0 | tile_dim0
    g1[4] = rows;                                // tile_dim1 | tile_dim2=0
    g1[5] = row_stride;                          // tensor_dim0_stride[31:0]
    g1[6] = 0u;
    g1[7] = 0u;
    asm volatile("tensor_load_to_lds %0, %1" :: "s"(g0), "s"(g1) : "memory");
}
#endif

// ---------------------------------------------------------------------------
// Main kernel: one workgroup (4 wave32) per (i,c) pair.
//   Phase 1: attn_raw (48x64, padded) = im_i x cap_c^T via
//            v_wmma_f32_16x16x32_f16; K chunks staged into double-buffered
//            LDS by the Tensor Data Mover, pipelined on TENSORcnt so the
//            DMA of chunk n+1 overlaps the WMMAs of chunk n.
//   Phase 2: leaky-relu, per-r norms, softmax over r, w12, Gram quadratic
//            form, masked LSE -> scores[i,c]. All in LDS.
// ---------------------------------------------------------------------------
__launch_bounds__(128)
__global__ void k_attn_scores(const half_t* __restrict__ imh, const half_t* __restrict__ caph,
                              const float* __restrict__ gram, const float* __restrict__ w1,
                              const int* __restrict__ s_l, float* __restrict__ scores) {
    constexpr int KC   = 128;        // K chunk (halves)
    constexpr int NCH  = Dn / KC;    // 8 chunks
    constexpr int ASTR = 136;        // LDS row stride (halves): 128 + 8 pad
    constexpr int BSTR = 136;
    constexpr int CSTR = 68;         // attn LDS row stride (floats)

    const int bx  = blockIdx.x;
    const int i   = bx >> 7;          // / 128
    const int c   = bx & 127;
    const int tid = threadIdx.x;
    const int lane = tid & 31;
    const int wv   = tid >> 5;        // wave id 0..3 -> N tile

    __shared__ __align__(16) half_t As[2][48 * ASTR];
    __shared__ __align__(16) half_t Bs[2][64 * BSTR];
    __shared__ float attnS[48 * CSTR];
    __shared__ float Gs[Rn * Rn];
    __shared__ float aL[Wn * 37];
    __shared__ float nrm[Rn];
    __shared__ float w12s[64];
    __shared__ float w2sq[64];
    __shared__ float ew[64];

    const half_t* Abase = imh  + (size_t)i * Rn * Dn;
    const half_t* Bbase = caph + (size_t)c * Wn * Dn;

    v8f acc0 = {}, acc1 = {}, acc2 = {};
    const int m0 = lane & 15;         // row/col within 16x16 tile
    const int hi = lane >> 4;         // lane half (0/1)
    const int n0 = wv * 16;           // this wave's N-tile base

    // Zero the constant pad rows of both buffers once (rows >= 36 of A,
    // >= 50 of B); staging only touches valid rows afterwards.
    for (int sgm = tid; sgm < 2 * 48 * (KC / 8); sgm += 128) {
        int b = sgm / (48 * (KC / 8)), s2 = sgm % (48 * (KC / 8));
        int r = s2 >> 4;
        if (r >= Rn) *(v8h*)(&As[b][r * ASTR + (s2 & 15) * 8]) = (v8h){};
    }
    for (int sgm = tid; sgm < 2 * 64 * (KC / 8); sgm += 128) {
        int b = sgm / (64 * (KC / 8)), s2 = sgm % (64 * (KC / 8));
        int r = s2 >> 4;
        if (r >= Wn) *(v8h*)(&Bs[b][r * BSTR + (s2 & 15) * 8]) = (v8h){};
    }

#if USE_TDM
    // prologue: DMA chunk 0 into buffer 0
    if (wv == 0) {
        tdm_load_2d((uint)(size_t)&As[0][0], Abase, Rn, Dn, KC);
        tdm_load_2d((uint)(size_t)&Bs[0][0], Bbase, Wn, Dn, KC);
    }
#endif

    for (int ch = 0; ch < NCH; ++ch) {
        const int cur = ch & 1;
        const int k0  = ch * KC;
#if USE_TDM
        if (wv == 0) {
            if (ch + 1 < NCH) {
                // prefetch next chunk into the other buffer (readers of that
                // buffer finished at the trailing barrier of iteration ch-1)
                tdm_load_2d((uint)(size_t)&As[1 - cur][0], Abase + k0 + KC, Rn, Dn, KC);
                tdm_load_2d((uint)(size_t)&Bs[1 - cur][0], Bbase + k0 + KC, Wn, Dn, KC);
                // tensor ops complete in order: <=2 outstanding means chunk
                // `ch`'s pair has landed while chunk ch+1 streams in.
                __builtin_amdgcn_s_wait_tensorcnt(2);
            } else {
                __builtin_amdgcn_s_wait_tensorcnt(0);
            }
        }
        __syncthreads();                       // buf[cur] ready for all waves
#else
        for (int sgm = tid; sgm < Rn * (KC / 8); sgm += 128) {
            int r = sgm >> 4, c8 = sgm & 15;
            *(v8h*)(&As[cur][r * ASTR + c8 * 8]) =
                *(const v8h*)(Abase + (size_t)r * Dn + k0 + c8 * 8);
        }
        for (int sgm = tid; sgm < Wn * (KC / 8); sgm += 128) {
            int r = sgm >> 4, c8 = sgm & 15;
            *(v8h*)(&Bs[cur][r * BSTR + c8 * 8]) =
                *(const v8h*)(Bbase + (size_t)r * Dn + k0 + c8 * 8);
        }
        __syncthreads();
#endif
        for (int kk = 0; kk < KC; kk += 32) {
            // B fragment (K x 16): lane N = lane&15; elems 0..15 -> K = kk + 16*hi + e
            union { v16h v; v8h h[2]; } bf;
            const half_t* bp = &Bs[cur][(n0 + m0) * BSTR + kk + hi * 16];
            bf.h[0] = *(const v8h*)bp;
            bf.h[1] = *(const v8h*)(bp + 8);
#pragma unroll
            for (int t = 0; t < 3; ++t) {
                // A fragment (16 x K): elems 0..15 -> K=kk+16*hi+e
                union { v16h v; v8h h[2]; } af;
                const half_t* ap = &As[cur][(t * 16 + m0) * ASTR + kk + hi * 16];
                af.h[0] = *(const v8h*)ap;
                af.h[1] = *(const v8h*)(ap + 8);
                v8f& accr = (t == 0) ? acc0 : ((t == 1) ? acc1 : acc2);
                accr = __builtin_amdgcn_wmma_f32_16x16x32_f16(
                    false, af.v, false, bf.v, (short)0, accr, false, false);
            }
        }
        __syncthreads();   // all waves done reading buf[cur]; it may be refilled
    }

    // write attn tiles to LDS: C layout VGPR j -> M = j (lanes 0-15) / 8+j (16-31)
#pragma unroll
    for (int t = 0; t < 3; ++t) {
        v8f a = (t == 0) ? acc0 : ((t == 1) ? acc1 : acc2);
#pragma unroll
        for (int j = 0; j < 8; ++j) {
            int m = t * 16 + hi * 8 + j;
            attnS[m * CSTR + n0 + m0] = a[j];
        }
    }
    __syncthreads();

    // stage Gram_i, zero w2sq, per-row leaky-relu L2 norms (over w)
    for (int t = tid; t < Rn * Rn; t += 128) Gs[t] = gram[(size_t)i * (Rn * Rn) + t];
    if (tid < 64) w2sq[tid] = 0.f;
    if (tid < Rn) {
        float ssum = 0.f;
        for (int w = 0; w < Wn; ++w) {
            float x = attnS[tid * CSTR + w];
            x = (x > 0.f) ? x : 0.1f * x;
            ssum += x * x;
        }
        nrm[tid] = sqrtf(ssum) + 1e-8f;
    }
    __syncthreads();

    // softmax over r for each w, plus w12[w] = sum_r a[w,r]*attn_raw[r,w]
    if (tid < Wn) {
        const int w = tid;
        float mx = -3.4e38f;
        for (int r = 0; r < Rn; ++r) {
            float x = attnS[r * CSTR + w];
            x = (x > 0.f) ? x : 0.1f * x;
            float lg = (x / nrm[r]) * 9.0f;
            aL[w * 37 + r] = lg;
            mx = fmaxf(mx, lg);
        }
        float se = 0.f;
        for (int r = 0; r < Rn; ++r) {
            float e = expf(aL[w * 37 + r] - mx);
            aL[w * 37 + r] = e;
            se += e;
        }
        float inv = 1.f / se;
        float w12 = 0.f;
        for (int r = 0; r < Rn; ++r) {
            float a = aL[w * 37 + r] * inv;
            aL[w * 37 + r] = a;
            w12 += a * attnS[r * CSTR + w];
        }
        w12s[w] = w12;
    }
    __syncthreads();

    // ||wei[w]||^2 = a^T Gram a, distributed over (w, r2) with LDS float atomics
    for (int e = tid; e < Wn * Rn; e += 128) {
        int w = e / Rn, r2 = e % Rn;
        float sres = 0.f;
        for (int r = 0; r < Rn; ++r) sres += aL[w * 37 + r] * Gs[r * Rn + r2];
        atomicAdd(&w2sq[w], sres * aL[w * 37 + r2]);
    }
    __syncthreads();

    if (tid < Wn) {
        const int w = tid;
        float w2  = sqrtf(w2sq[w]);
        float w1v = w1[c * Wn + w];
        float sim = w12s[w] / fmaxf(w1v * w2, 1e-8f);
        ew[w] = (w < s_l[c]) ? expf(sim * 6.0f) : 0.f;
    }
    __syncthreads();

    if (tid == 0) {
        float ssum = 0.f;
        for (int w = 0; w < Wn; ++w) ssum += ew[w];
        scores[i * Bn + c] = logf(ssum) / 6.0f;
    }
}

// ---------------------------------------------------------------------------
// Final hinge loss reduction over the 128x128 score matrix
// ---------------------------------------------------------------------------
__global__ void k_loss(const float* __restrict__ scores, float* __restrict__ out) {
    __shared__ float red[Bn];
    int t = threadIdx.x;
    float dt = scores[t * Bn + t];
    float mS = 0.f, mI = 0.f;   // relu(.) and eye->0 are absorbed by the 0 init
    for (int k = 0; k < Bn; ++k) {
        if (k == t) continue;
        mS = fmaxf(mS, 0.2f + scores[t * Bn + k] - dt);  // cost_s row t (diag[t])
        mI = fmaxf(mI, 0.2f + scores[k * Bn + t] - dt);  // cost_im col t (diag[t])
    }
    red[t] = mS + mI;
    __syncthreads();
    if (t == 0) {
        float ssum = 0.f;
        for (int k = 0; k < Bn; ++k) ssum += red[k];
        out[0] = ssum;
    }
}

// ---------------------------------------------------------------------------
extern "C" void kernel_launch(void* const* d_in, const int* in_sizes, int n_in,
                              void* d_out, int out_size, void* d_ws, size_t ws_size,
                              hipStream_t stream) {
    (void)in_sizes; (void)n_in; (void)out_size; (void)ws_size;
    const float* im  = (const float*)d_in[0];
    // d_in[1] = im_l : always R, unused by the math
    const float* s   = (const float*)d_in[2];
    const int*   s_l = (const int*)d_in[3];

    char* p = (char*)d_ws;
    auto alloc = [&](size_t bytes) -> char* {
        char* r = p;
        p += (bytes + 255) & ~(size_t)255;
        return r;
    };
    half_t* imh    = (half_t*)alloc((size_t)Bn * Rn * Dn * sizeof(half_t));
    half_t* caph   = (half_t*)alloc((size_t)Bn * Wn * Dn * sizeof(half_t));
    float*  w1     = (float*) alloc((size_t)Bn * Wn * sizeof(float));
    float*  gram   = (float*) alloc((size_t)Bn * Rn * Rn * sizeof(float));
    float*  scores = (float*) alloc((size_t)Bn * Bn * sizeof(float));

    int nim = Bn * Rn * Dn;
    k_prep_im<<<dim3((nim + 255) / 256), dim3(256), 0, stream>>>(im, imh, nim);
    k_prep_cap<<<dim3(Bn * Wn), dim3(128), 0, stream>>>(s, s_l, caph, w1);
    k_prep_gram<<<dim3(Bn), dim3(256), 0, stream>>>(im, gram);
    k_attn_scores<<<dim3(Bn * Bn), dim3(128), 0, stream>>>(imh, caph, gram, w1, s_l, scores);
    k_loss<<<dim3(1), dim3(128), 0, stream>>>(scores, (float*)d_out);
}
